// OptionCompareCell_3942779978094
// MI455X (gfx1250) — compile-verified, matrix operand order checked by reading, not run on hardware
//
#include <hip/hip_runtime.h>
#include <string.h>
#include <math.h>

// ---------------------------------------------------------------------------
// OptionCompareCell for MI455X (gfx1250, wave32, WMMA bf16 16x16x32)
// - all heavy GEMMs on v_wmma_f32_16x16x32_bf16, f32 accumulation
// - ctx_sum linearity trick removes the per-pair OLK GEMM (116 GF -> 38 GF)
// - weights pre-transposed to [N,K] so staging is vector b128 loads
// - LDS tiles double-buffered; staged with GLOBAL_LOAD_ASYNC_TO_LDS_B128
//   (ASYNCcnt) when the builtin is available, sync b128 copies otherwise
// ---------------------------------------------------------------------------

typedef __attribute__((ext_vector_type(16))) __bf16 v16bf;
typedef __attribute__((ext_vector_type(8)))  float  v8f;
typedef int v4i __attribute__((vector_size(16)));   // async-lds builtin ptr type

#if defined(__HIP_DEVICE_COMPILE__)
#if __has_builtin(__builtin_amdgcn_global_load_async_to_lds_b128) && \
    __has_builtin(__builtin_amdgcn_s_wait_asynccnt)
#define USE_ASYNC_LDS 1
#endif
#endif
#ifndef USE_ASYNC_LDS
#define USE_ASYNC_LDS 0
#endif

static constexpr int Bb  = 64, Cc = 4, LOo = 64, LQq = 128, Hh = 768;
static constexpr int BC  = Bb * Cc;            // 256
static constexpr int BP  = Bb * Cc * (Cc - 1); // 768 (b, i, j!=i)
static constexpr int RO  = BC * LOo;           // 16384 option-token rows
static constexpr int RQ  = BC * LQq;           // 32768 question-token rows
static constexpr float NEGV = -10000.0f;

__device__ __forceinline__ unsigned short f2bf(float x) {
    unsigned int u = __float_as_uint(x);
    u += 0x7FFFu + ((u >> 16) & 1u);           // round-to-nearest-even
    return (unsigned short)(u >> 16);
}

union FragU { uint4 q[2]; v16bf v; };
__device__ __forceinline__ v16bf ldsFrag(const unsigned short* p) {
    FragU f;
    const uint4* q = (const uint4*)p;
    f.q[0] = q[0];
    f.q[1] = q[1];
    return f.v;
}

// ---------------------------------------------------------------------------
// Generic bf16 WMMA GEMM: out[bz] = A[aMap(bz)] x B[bMap(bz)] (+bias/row/col)
// Block tile 64x64, BK=32, 8 waves (4x2), each wave 16x32 (2 wmma tiles).
// A memory: [M,K] row-major (lda). B memory: [N,K] (BNK=1) or [K,N] (BNK=0).
// Epilogues: 0 none, 1 tanh(x+add1), 2 merge g=sig(x); g*add1+(1-g)*add2,
//            3 relu, 4 atomicAdd into outF.
// Optional bf16 store into a (possibly wider) concat buffer at colOff.
// ---------------------------------------------------------------------------
struct GemmP {
    const unsigned short* A; const unsigned short* B;
    const int* aMap; const int* bMap; const int* oMap;
    const float* bias; const float* rowAdd; const float* colAdd;
    const float* add1; const float* add2;
    float* outF; unsigned short* outB;
    long long aBS, bBS, addBS, oFBS, oBBS;
    int lda, ldb, M, N, K;
    int rowBS, colBS, ldAdd, ldoF, ldoB, colOff, epi;
    float biasScale;
};

template <int BNK>
__global__ __launch_bounds__(256) void gemm_bf16_wmma(GemmP p) {
    __shared__ __align__(16) unsigned short As[2][64 * 32];
    __shared__ __align__(16) unsigned short Bs[2][64 * 32];

    const int tid  = threadIdx.x;
    const int lane = tid & 31;
    const int wid  = tid >> 5;
    const int wm   = wid & 3;       // wave row (of 4)
    const int wn   = wid >> 2;      // wave col (of 2)
    const int bz   = blockIdx.z;
    const int aB   = p.aMap ? p.aMap[bz] : bz;
    const int bB   = p.bMap ? p.bMap[bz] : bz;
    const int oB   = p.oMap ? p.oMap[bz] : bz;

    const unsigned short* Ag = p.A + (long long)aB * p.aBS
                                   + (long long)(blockIdx.x * 64) * p.lda;
    const unsigned short* Bg = p.B + (long long)bB * p.bBS;
    const int n0 = blockIdx.y * 64;

    // staging geometry (all in-loop offsets are 32-bit)
    const int sr    = tid >> 2;          // row 0..63
    const int sc    = (tid & 3) * 8;     // 8 bf16 = one b128
    const int aOff0 = sr * p.lda + sc;
    const int bOff1 = (n0 + sr) * p.ldb + sc;   // BNK=1 path
    const int bCol  = n0 + (tid & 63);          // BNK=0 path
    const int bRow0 = (tid >> 6) * 8;

    auto stage = [&](int k0, int buf) {
#if USE_ASYNC_LDS
        __builtin_amdgcn_global_load_async_to_lds_b128(
            (v4i*)(Ag + aOff0 + k0),
            (v4i*)&As[buf][sr * 32 + sc],
            0, 0);
#else
        *(uint4*)&As[buf][sr * 32 + sc] = *(const uint4*)(Ag + aOff0 + k0);
#endif
        if constexpr (BNK) {
#if USE_ASYNC_LDS
            __builtin_amdgcn_global_load_async_to_lds_b128(
                (v4i*)(Bg + bOff1 + k0),
                (v4i*)&Bs[buf][sr * 32 + sc],
                0, 0);
#else
            *(uint4*)&Bs[buf][sr * 32 + sc] = *(const uint4*)(Bg + bOff1 + k0);
#endif
        } else {
            const unsigned short* cp = Bg + (k0 + bRow0) * p.ldb + bCol;
            int off = 0;
#pragma unroll
            for (int r = 0; r < 8; ++r) {
                Bs[buf][(tid & 63) * 32 + bRow0 + r] = cp[off];
                off += p.ldb;
            }
        }
    };

    v8f acc0 = {};
    v8f acc1 = {};
    stage(0, 0);
    int buf = 0;
    for (int k0 = 0; k0 < p.K; k0 += 32) {
#if USE_ASYNC_LDS
        __builtin_amdgcn_s_wait_asynccnt(0);
#endif
        __syncthreads();
        if (k0 + 32 < p.K) stage(k0 + 32, buf ^ 1);   // overlap next tile
        const int aRow = wm * 16 + (lane & 15);
        const int kh   = (lane >> 4) * 16;
        v16bf a  = ldsFrag(&As[buf][aRow * 32 + kh]);
        v16bf b0 = ldsFrag(&Bs[buf][(wn * 32 + (lane & 15)) * 32 + kh]);
        v16bf b1 = ldsFrag(&Bs[buf][(wn * 32 + 16 + (lane & 15)) * 32 + kh]);
        acc0 = __builtin_amdgcn_wmma_f32_16x16x32_bf16(
                   false, a, false, b0, (short)0, acc0, false, false);
        acc1 = __builtin_amdgcn_wmma_f32_16x16x32_bf16(
                   false, a, false, b1, (short)0, acc1, false, false);
        buf ^= 1;
    }

    // epilogue: c/d layout -> lanes 0-15: m=r, n=lane; lanes 16-31: m=r+8
    const int mBase = blockIdx.x * 64 + wm * 16 + ((lane >> 4) << 3);
#pragma unroll
    for (int t = 0; t < 2; ++t) {
        v8f acc = t ? acc1 : acc0;
        const int n = n0 + wn * 32 + t * 16 + (lane & 15);
#pragma unroll
        for (int r = 0; r < 8; ++r) {
            const int m = mBase + r;
            float v = acc[r];
            if (p.bias)   v += p.bias[n] * p.biasScale;
            if (p.rowAdd) v += p.rowAdd[(long long)aB * p.rowBS + m];
            if (p.colAdd) v += p.colAdd[(long long)bB * p.colBS + n];
            if (p.epi == 1 || p.epi == 2) {
                const long long ai = (long long)oB * p.addBS
                                   + (long long)m * p.ldAdd + n;
                if (p.epi == 1) {
                    v = tanhf(v + p.add1[ai]);
                } else {
                    const float g = 1.0f / (1.0f + __expf(-v));
                    v = g * p.add1[ai] + (1.0f - g) * p.add2[ai];
                }
            } else if (p.epi == 3) {
                v = fmaxf(v, 0.0f);
            }
            if (p.outF) {
                float* o = p.outF + (long long)oB * p.oFBS
                                  + (long long)m * p.ldoF + n;
                if (p.epi == 4) atomicAdd(o, v); else *o = v;
            }
            if (p.outB)
                p.outB[(long long)oB * p.oBBS + (long long)m * p.ldoB
                       + p.colOff + n] = f2bf(v);
        }
    }
}

// ---------------------------------------------------------------------------
// Small kernels
// ---------------------------------------------------------------------------
__global__ void k_fill0(float* p, long long n) {
    long long i = (long long)blockIdx.x * blockDim.x + threadIdx.x;
    const long long s = (long long)gridDim.x * blockDim.x;
    for (; i < n; i += s) p[i] = 0.0f;
}

__global__ void k_maps(int* i2bc, int* j2bc) {
    const int bp = blockIdx.x * blockDim.x + threadIdx.x;
    if (bp >= BP) return;
    const int b  = bp / 12, pr = bp % 12;
    const int i  = pr / 3,  jj = pr % 3;
    const int j  = jj + (jj >= i ? 1 : 0);
    i2bc[bp] = b * Cc + i;
    j2bc[bp] = b * Cc + j;
}

// f32 -> bf16, optional per-feature scale vector (length Hh)
__global__ void k_cvt_bf(const float* x, const float* scale,
                         unsigned short* y, long long n) {
    long long i = (long long)blockIdx.x * blockDim.x + threadIdx.x;
    const long long s = (long long)gridDim.x * blockDim.x;
    for (; i < n; i += s) {
        float v = x[i];
        if (scale) v *= scale[(int)(i % Hh)];
        y[i] = f2bf(v);
    }
}

// f32 [K,N] -> bf16 [N,K] (weight pre-transpose for fast b128 staging)
__global__ void k_cvt_bfT(const float* x, unsigned short* y, int K, int N) {
    long long i = (long long)blockIdx.x * blockDim.x + threadIdx.x;
    const long long s = (long long)gridDim.x * blockDim.x;
    const long long n = (long long)K * N;
    for (; i < n; i += s) {
        const int k = (int)(i / N), c = (int)(i % N);
        y[(long long)c * K + k] = f2bf(x[i]);
    }
}

// y[r] = X[r,:]@w + bias[0] + (1-mask[r])*NEG   (wave per row)
__global__ __launch_bounds__(256) void k_rowproj(const float* X, const float* w,
                                                 const float* bias,
                                                 const int* mask, float* out,
                                                 int rows) {
    const int row  = blockIdx.x * 8 + (threadIdx.x >> 5);
    const int lane = threadIdx.x & 31;
    if (row >= rows) return;
    const float* xr = X + (long long)row * Hh;
    float s = 0.0f;
    for (int d = lane; d < Hh; d += 32) s += xr[d] * w[d];
    for (int off = 16; off; off >>= 1) s += __shfl_xor(s, off);
    if (lane == 0) {
        if (bias) s += bias[0];
        if (mask) s += (1.0f - (float)mask[row]) * NEGV;
        out[row] = s;
    }
}

// softmax over axis-1 (U) of in[Nb,U,V]; bf16 TRANSPOSED out [Nb,V,U]
__global__ __launch_bounds__(256) void k_softmax_ax1T(const float* in,
                                                      unsigned short* out,
                                                      int Nb, int U, int V) {
    const int idx  = blockIdx.x * 8 + (threadIdx.x >> 5);
    const int lane = threadIdx.x & 31;
    if (idx >= Nb * V) return;
    const int nb = idx / V, v = idx % V;
    const float* col = in + (long long)nb * U * V + v;
    float vals[4];
    int   cnt = 0;
    float mx  = -1e30f;
    for (int u = lane; u < U; u += 32) {
        const float x = col[(long long)u * V];
        vals[cnt++] = x;
        mx = fmaxf(mx, x);
    }
    for (int off = 16; off; off >>= 1) mx = fmaxf(mx, __shfl_xor(mx, off));
    float s = 0.0f;
    for (int i = 0; i < cnt; ++i) { vals[i] = __expf(vals[i] - mx); s += vals[i]; }
    for (int off = 16; off; off >>= 1) s += __shfl_xor(s, off);
    const float inv = 1.0f / s;
    unsigned short* orow = out + (long long)nb * V * U + (long long)v * U;
    cnt = 0;
    for (int u = lane; u < U; u += 32) orow[u] = f2bf(vals[cnt++] * inv);
}

// softmax over axis-2 (V) of in[Nb,U,V]; bf16 row-major out
__global__ __launch_bounds__(256) void k_softmax_ax2(const float* in,
                                                     unsigned short* out,
                                                     int Nb, int U, int V) {
    const int idx  = blockIdx.x * 8 + (threadIdx.x >> 5);
    const int lane = threadIdx.x & 31;
    if (idx >= Nb * U) return;
    const float* row = in + (long long)idx * V;
    float vals[4];
    int   cnt = 0;
    float mx  = -1e30f;
    for (int v = lane; v < V; v += 32) {
        const float x = row[v];
        vals[cnt++] = x;
        mx = fmaxf(mx, x);
    }
    for (int off = 16; off; off >>= 1) mx = fmaxf(mx, __shfl_xor(mx, off));
    float s = 0.0f;
    for (int i = 0; i < cnt; ++i) { vals[i] = __expf(vals[i] - mx); s += vals[i]; }
    for (int off = 16; off; off >>= 1) s += __shfl_xor(s, off);
    const float inv = 1.0f / s;
    unsigned short* orow = out + (long long)idx * V;
    cnt = 0;
    for (int v = lane; v < V; v += 32) orow[v] = f2bf(vals[cnt++] * inv);
}

// per-bc question pooling: Aq = softmax(sVa[bc]), Q[bc,d] = sum_l Aq[l]*q[l,d]
__global__ __launch_bounds__(256) void k_aqQ(const float* sVa, const float* q,
                                             float* Q) {
    const int bc  = blockIdx.x;
    const int tid = threadIdx.x;
    __shared__ float aq[LQq];
    __shared__ float stat[2];
    if (tid < LQq) aq[tid] = sVa[bc * LQq + tid];
    __syncthreads();
    if (tid == 0) {
        float mx = -1e30f;
        for (int l = 0; l < LQq; ++l) mx = fmaxf(mx, aq[l]);
        float s = 0.0f;
        for (int l = 0; l < LQq; ++l) s += __expf(aq[l] - mx);
        stat[0] = mx;
        stat[1] = 1.0f / s;
    }
    __syncthreads();
    if (tid < LQq) aq[tid] = __expf(aq[tid] - stat[0]) * stat[1];
    __syncthreads();
    const float* qb = q + (long long)bc * LQq * Hh;
    for (int d = tid; d < Hh; d += 256) {
        float acc = 0.0f;
        for (int l = 0; l < LQq; ++l) acc += qb[(long long)l * Hh + d] * aq[l];
        Q[(long long)bc * Hh + d] = acc;
    }
}

// S = [3*o - ctx, o*ctx]  (bf16, [RO, 2H])
__global__ void k_buildS(const float* o, const float* ctx,
                         unsigned short* S, long long n) {
    long long i = (long long)blockIdx.x * blockDim.x + threadIdx.x;
    const long long s = (long long)gridDim.x * blockDim.x;
    for (; i < n; i += s) {
        const long long row = i / Hh;
        const int       d   = (int)(i % Hh);
        const float ov = o[i], cv = ctx[i];
        S[row * 2 * Hh + d]      = f2bf(3.0f * ov - cv);
        S[row * 2 * Hh + Hh + d] = f2bf(ov * cv);
    }
}

// Gcat = [o, OCK, Q(broadcast over lo)]  (bf16, [RO, 3H])
__global__ void k_buildGcat(const float* o, const float* ock, const float* Q,
                            unsigned short* G, long long n) {
    long long i = (long long)blockIdx.x * blockDim.x + threadIdx.x;
    const long long s = (long long)gridDim.x * blockDim.x;
    for (; i < n; i += s) {
        const long long row = i / Hh;
        const int       d   = (int)(i % Hh);
        const long long bc  = row >> 6;   // LO=64
        G[row * 3 * Hh + d]          = f2bf(o[i]);
        G[row * 3 * Hh + Hh + d]     = f2bf(ock[i]);
        G[row * 3 * Hh + 2 * Hh + d] = f2bf(Q[bc * Hh + d]);
    }
}

// strided bf16 copy (q_bf -> dOA[:, :, 0:H])
__global__ void k_copy_bf(const unsigned short* src, unsigned short* dst,
                          long long n, int w, int ldd, int colOff) {
    long long i = (long long)blockIdx.x * blockDim.x + threadIdx.x;
    const long long s = (long long)gridDim.x * blockDim.x;
    for (; i < n; i += s) {
        const long long row = i / w;
        const int       c   = (int)(i % w);
        dst[row * ldd + colOff + c] = src[i];
    }
}

// Fcat = [OPK, OSK, OPK-OSK, OPK*OSK]  (bf16, [RO, 4H])
__global__ void k_buildFcat(const float* p, const float* s2,
                            unsigned short* F, long long n) {
    long long i = (long long)blockIdx.x * blockDim.x + threadIdx.x;
    const long long s = (long long)gridDim.x * blockDim.x;
    for (; i < n; i += s) {
        const long long row = i / Hh;
        const int       d   = (int)(i % Hh);
        const float pv = p[i], sv = s2[i];
        F[row * 4 * Hh + d]          = f2bf(pv);
        F[row * 4 * Hh + Hh + d]     = f2bf(sv);
        F[row * 4 * Hh + 2 * Hh + d] = f2bf(pv - sv);
        F[row * 4 * Hh + 3 * Hh + d] = f2bf(pv * sv);
    }
}

// out[bc,h] = max over lo of OFK[bc,lo,h]
__global__ void k_maxred(const float* OFK, float* out) {
    const int i = blockIdx.x * blockDim.x + threadIdx.x;
    if (i >= BC * Hh) return;
    const int bc = i / Hh, d = i % Hh;
    const float* p = OFK + (long long)bc * LOo * Hh + d;
    float m = p[0];
    for (int l = 1; l < LOo; ++l) m = fmaxf(m, p[(long long)l * Hh]);
    out[i] = m;
}

// ---------------------------------------------------------------------------
// Host orchestration
// ---------------------------------------------------------------------------
static inline GemmP gp0() {
    GemmP p;
    memset(&p, 0, sizeof(p));
    p.biasScale = 1.0f;
    return p;
}

extern "C" void kernel_launch(void* const* d_in, const int* in_sizes, int n_in,
                              void* d_out, int out_size, void* d_ws,
                              size_t ws_size, hipStream_t stream) {
    const float* o_f  = (const float*)d_in[0];
    const float* q_f  = (const float*)d_in[1];
    const int*   om   = (const int*)d_in[2];
    const int*   qm   = (const int*)d_in[3];
    const float* attW1 = (const float*)d_in[4];  const float* attW1b = (const float*)d_in[5];
    const float* attW2 = (const float*)d_in[6];  const float* attW2b = (const float*)d_in[7];
    const float* attW3 = (const float*)d_in[8];
    const float* WcS   = (const float*)d_in[9];  const float* WcSb   = (const float*)d_in[10];
    const float* Wc    = (const float*)d_in[11]; const float* Wcb    = (const float*)d_in[12];
    const float* Va    = (const float*)d_in[13]; const float* Vab    = (const float*)d_in[14];
    const float* Wg    = (const float*)d_in[15]; const float* Wgb    = (const float*)d_in[16];
    const float* coW1  = (const float*)d_in[17]; const float* coW1b  = (const float*)d_in[18];
    const float* coW2  = (const float*)d_in[19]; const float* coW2b  = (const float*)d_in[20];
    const float* coW3  = (const float*)d_in[21];
    const float* Wp    = (const float*)d_in[22]; const float* Wpb    = (const float*)d_in[23];
    const float* saW1  = (const float*)d_in[24]; const float* saW1b  = (const float*)d_in[25];
    const float* saW2  = (const float*)d_in[26]; const float* saW2b  = (const float*)d_in[27];
    const float* saW3  = (const float*)d_in[28];
    const float* Wf    = (const float*)d_in[29]; const float* Wfb    = (const float*)d_in[30];
    (void)in_sizes; (void)n_in; (void)out_size;

    char*  W   = (char*)d_ws;
    size_t cur = 0;
    auto alloc = [&](size_t bytes) -> void* {
        void* p = W + cur;
        cur = (cur + bytes + 255) & ~(size_t)255;
        return p;
    };
    typedef unsigned short u16;
    // bf16 activations / transposed weights
    u16* o_bf    = (u16*)alloc((size_t)RO * Hh * 2);
    u16* q_bf    = (u16*)alloc((size_t)RQ * Hh * 2);
    u16* ow3_bf  = (u16*)alloc((size_t)RO * Hh * 2);
    u16* qw3_bf  = (u16*)alloc((size_t)RQ * Hh * 2);
    u16* opw3_bf = (u16*)alloc((size_t)RO * Hh * 2);
    u16* wcsT    = (u16*)alloc((size_t)Hh * Hh * 2);
    u16* wcT     = (u16*)alloc((size_t)2 * Hh * Hh * 2);
    u16* wgT     = (u16*)alloc((size_t)3 * Hh * Hh * 2);
    u16* wpT     = (u16*)alloc((size_t)3 * Hh * Hh * 2);
    u16* wfT     = (u16*)alloc((size_t)4 * Hh * Hh * 2);
    // stage 1
    float* pv1a  = (float*)alloc((size_t)RO * 4);
    float* pv2a  = (float*)alloc((size_t)RO * 4);
    float* A1    = (float*)alloc((size_t)BP * LOo * LOo * 4);   // reused: A2, A3
    u16*  attT   = (u16*)alloc((size_t)BP * LOo * LOo * 2);
    float* ctx   = (float*)alloc((size_t)RO * Hh * 4);          // reused: OSK
    u16*  S_bf   = (u16*)alloc((size_t)RO * 2 * Hh * 2);
    float* cmp   = (float*)alloc((size_t)RO * Hh * 4);          // reused: OFK
    // merge
    float* sVa   = (float*)alloc((size_t)RQ * 4);
    float* Qv    = (float*)alloc((size_t)BC * Hh * 4);
    float* OCK   = (float*)alloc((size_t)RO * Hh * 4);
    u16*  Gcat   = (u16*)alloc((size_t)RO * 3 * Hh * 2);
    float* mrg   = (float*)alloc((size_t)RO * Hh * 4);
    u16*  Pcat   = (u16*)alloc((size_t)RO * 3 * Hh * 2);
    // co-attention
    float* pq    = (float*)alloc((size_t)RQ * 4);
    float* pm    = (float*)alloc((size_t)RO * 4);
    u16*  ACK    = (u16*)alloc((size_t)BC * LQq * LOo * 2);
    u16*  APKT   = (u16*)alloc((size_t)BC * LQq * LOo * 2);
    u16*  dOA    = (u16*)alloc((size_t)RQ * 2 * Hh * 2);        // reused: Fcat
    float* OPKf  = (float*)alloc((size_t)RO * Hh * 4);
    u16*  OPKb   = (u16*)alloc((size_t)RO * Hh * 2);
    float* pv1s  = (float*)alloc((size_t)RO * 4);
    float* pv2s  = (float*)alloc((size_t)RO * 4);
    u16*  at3T   = (u16*)alloc((size_t)BC * LOo * LOo * 2);
    int*  i2bc   = (int*)alloc((size_t)BP * 4);
    int*  j2bc   = (int*)alloc((size_t)BP * 4);
    float* A2    = A1;
    float* A3    = A1;
    float* OSK   = ctx;
    float* OFK   = cmp;
    u16*  Fcat   = dOA;
    (void)ws_size;

    float* outp = (float*)d_out;

    auto eblocks = [](long long n) { return (int)((n + 255) / 256); };
    auto gemm = [&](const GemmP& p, int batch, int bNK) {
        dim3 g(p.M / 64, p.N / 64, batch);
        if (bNK) gemm_bf16_wmma<1><<<g, dim3(256), 0, stream>>>(p);
        else     gemm_bf16_wmma<0><<<g, dim3(256), 0, stream>>>(p);
    };

    // ---- precompute -------------------------------------------------------
    k_maps<<<dim3(3), dim3(256), 0, stream>>>(i2bc, j2bc);
    k_cvt_bf<<<eblocks((long long)RO * Hh), 256, 0, stream>>>(o_f, nullptr, o_bf, (long long)RO * Hh);
    k_cvt_bf<<<eblocks((long long)RQ * Hh), 256, 0, stream>>>(q_f, nullptr, q_bf, (long long)RQ * Hh);
    k_cvt_bf<<<eblocks((long long)RO * Hh), 256, 0, stream>>>(o_f, attW3, ow3_bf, (long long)RO * Hh);
    k_cvt_bf<<<eblocks((long long)RQ * Hh), 256, 0, stream>>>(q_f, coW3, qw3_bf, (long long)RQ * Hh);
    k_cvt_bfT<<<eblocks((long long)Hh * Hh), 256, 0, stream>>>(WcS, wcsT, Hh, Hh);
    k_cvt_bfT<<<eblocks((long long)2 * Hh * Hh), 256, 0, stream>>>(Wc, wcT, 2 * Hh, Hh);
    k_cvt_bfT<<<eblocks((long long)3 * Hh * Hh), 256, 0, stream>>>(Wg, wgT, 3 * Hh, Hh);
    k_cvt_bfT<<<eblocks((long long)3 * Hh * Hh), 256, 0, stream>>>(Wp, wpT, 3 * Hh, Hh);
    k_cvt_bfT<<<eblocks((long long)4 * Hh * Hh), 256, 0, stream>>>(Wf, wfT, 4 * Hh, Hh);
    k_rowproj<<<(RO + 7) / 8, 256, 0, stream>>>(o_f, attW1, attW1b, om, pv1a, RO);
    k_rowproj<<<(RO + 7) / 8, 256, 0, stream>>>(o_f, attW2, attW2b, om, pv2a, RO);
    k_fill0<<<eblocks((long long)RO * Hh), 256, 0, stream>>>(ctx, (long long)RO * Hh);

    // ---- stage 1: pairwise compare ---------------------------------------
    { // A1[bp,u,v] = (o_j*w3) @ o_i^T + p1_j[u] + p2_i[v]
        GemmP p = gp0();
        p.A = ow3_bf; p.aBS = (long long)LOo * Hh; p.aMap = j2bc; p.lda = Hh;
        p.B = o_bf;   p.bBS = (long long)LOo * Hh; p.bMap = i2bc; p.ldb = Hh;
        p.M = LOo; p.N = LOo; p.K = Hh;
        p.rowAdd = pv1a; p.rowBS = LOo; p.colAdd = pv2a; p.colBS = LOo;
        p.outF = A1; p.oFBS = (long long)LOo * LOo; p.ldoF = LOo;
        gemm(p, BP, 1);
    }
    k_softmax_ax1T<<<(BP * LOo + 7) / 8, 256, 0, stream>>>(A1, attT, BP, LOo, LOo);
    { // ctx_sum[b,i] += att^T @ o_j   (atomic-add epilogue, 3 pairs per (b,i))
        GemmP p = gp0();
        p.A = attT; p.aBS = (long long)LOo * LOo; p.lda = LOo;
        p.B = o_bf; p.bBS = (long long)LOo * Hh; p.bMap = j2bc; p.ldb = Hh;
        p.M = LOo; p.N = Hh; p.K = LOo;
        p.outF = ctx; p.oFBS = (long long)LOo * Hh; p.ldoF = Hh; p.oMap = i2bc;
        p.epi = 4;
        gemm(p, BP, 0);
    }
    k_buildS<<<eblocks((long long)RO * Hh), 256, 0, stream>>>(o_f, ctx, S_bf, (long long)RO * Hh);
    { // cmp_sum = S @ Wc + 3*Wc_b
        GemmP p = gp0();
        p.A = S_bf; p.lda = 2 * Hh;
        p.B = wcT; p.ldb = 2 * Hh;
        p.M = RO; p.N = Hh; p.K = 2 * Hh;
        p.bias = Wcb; p.biasScale = 3.0f;
        p.outF = cmp; p.ldoF = Hh;
        gemm(p, 1, 1);
    }

    // ---- merge layer ------------------------------------------------------
    k_rowproj<<<(RQ + 7) / 8, 256, 0, stream>>>(q_f, Va, Vab, qm, sVa, RQ);
    k_aqQ<<<dim3(BC), dim3(256), 0, stream>>>(sVa, q_f, Qv);
    { // OCK = tanh(o @ WcSelf + b + cmp_sum)
        GemmP p = gp0();
        p.A = o_bf; p.lda = Hh;
        p.B = wcsT; p.ldb = Hh;
        p.M = RO; p.N = Hh; p.K = Hh;
        p.bias = WcSb;
        p.add1 = cmp; p.ldAdd = Hh; p.epi = 1;
        p.outF = OCK; p.ldoF = Hh;
        gemm(p, 1, 1);
    }
    k_buildGcat<<<eblocks((long long)RO * Hh), 256, 0, stream>>>(o_f, OCK, Qv, Gcat, (long long)RO * Hh);
    { // merged = sig(Gcat@Wg+b)*o + (1-sig)*OCK -> f32 + bf16 into Pcat[:,0:H]
        GemmP p = gp0();
        p.A = Gcat; p.lda = 3 * Hh;
        p.B = wgT; p.ldb = 3 * Hh;
        p.M = RO; p.N = Hh; p.K = 3 * Hh;
        p.bias = Wgb;
        p.add1 = o_f; p.add2 = OCK; p.ldAdd = Hh; p.epi = 2;
        p.outF = mrg; p.ldoF = Hh;
        p.outB = Pcat; p.ldoB = 3 * Hh; p.colOff = 0;
        gemm(p, 1, 1);
    }

    // ---- co-attention with question --------------------------------------
    k_rowproj<<<(RQ + 7) / 8, 256, 0, stream>>>(q_f, coW1, coW1b, qm, pq, RQ);
    k_rowproj<<<(RO + 7) / 8, 256, 0, stream>>>(mrg, coW2, coW2b, om, pm, RO);
    { // A2[bc,q,o] = (q*w3) @ merged^T + pq[q] + pm[o]
        GemmP p = gp0();
        p.A = qw3_bf; p.aBS = (long long)LQq * Hh; p.lda = Hh;
        p.B = Pcat; p.bBS = (long long)LOo * 3 * Hh; p.ldb = 3 * Hh;
        p.M = LQq; p.N = LOo; p.K = Hh;
        p.rowAdd = pq; p.rowBS = LQq; p.colAdd = pm; p.colBS = LOo;
        p.outF = A2; p.oFBS = (long long)LQq * LOo; p.ldoF = LOo;
        gemm(p, BC, 1);
    }
    k_softmax_ax2<<<(BC * LQq + 7) / 8, 256, 0, stream>>>(A2, ACK, BC, LQq, LOo);
    k_softmax_ax1T<<<(BC * LOo + 7) / 8, 256, 0, stream>>>(A2, APKT, BC, LQq, LOo);
    k_copy_bf<<<eblocks((long long)RQ * Hh), 256, 0, stream>>>(q_bf, dOA, (long long)RQ * Hh, Hh, 2 * Hh, 0);
    { // OA = ACK @ merged -> bf16 into dOA[:, :, H:2H]
        GemmP p = gp0();
        p.A = ACK; p.aBS = (long long)LQq * LOo; p.lda = LOo;
        p.B = Pcat; p.bBS = (long long)LOo * 3 * Hh; p.ldb = 3 * Hh;
        p.M = LQq; p.N = Hh; p.K = LOo;
        p.outB = dOA; p.oBBS = (long long)LQq * 2 * Hh; p.ldoB = 2 * Hh; p.colOff = Hh;
        gemm(p, BC, 0);
    }
    { // POAA = APK^T @ dOA -> bf16 into Pcat[:, :, H:3H]
        GemmP p = gp0();
        p.A = APKT; p.aBS = (long long)LOo * LQq; p.lda = LQq;
        p.B = dOA; p.bBS = (long long)LQq * 2 * Hh; p.ldb = 2 * Hh;
        p.M = LOo; p.N = 2 * Hh; p.K = LQq;
        p.outB = Pcat; p.oBBS = (long long)LOo * 3 * Hh; p.ldoB = 3 * Hh; p.colOff = Hh;
        gemm(p, BC, 0);
    }
    { // OPK = relu(Pcat @ Wp + b)
        GemmP p = gp0();
        p.A = Pcat; p.lda = 3 * Hh;
        p.B = wpT; p.ldb = 3 * Hh;
        p.M = RO; p.N = Hh; p.K = 3 * Hh;
        p.bias = Wpb; p.epi = 3;
        p.outF = OPKf; p.ldoF = Hh;
        p.outB = OPKb; p.ldoB = Hh;
        gemm(p, 1, 1);
    }

    // ---- self-attention ---------------------------------------------------
    k_rowproj<<<(RO + 7) / 8, 256, 0, stream>>>(OPKf, saW1, saW1b, om, pv1s, RO);
    k_rowproj<<<(RO + 7) / 8, 256, 0, stream>>>(OPKf, saW2, saW2b, om, pv2s, RO);
    k_cvt_bf<<<eblocks((long long)RO * Hh), 256, 0, stream>>>(OPKf, saW3, opw3_bf, (long long)RO * Hh);
    { // A3[bc,u,v] = (OPK*w3) @ OPK^T + p1[u] + p2[v]
        GemmP p = gp0();
        p.A = opw3_bf; p.aBS = (long long)LOo * Hh; p.lda = Hh;
        p.B = OPKb; p.bBS = (long long)LOo * Hh; p.ldb = Hh;
        p.M = LOo; p.N = LOo; p.K = Hh;
        p.rowAdd = pv1s; p.rowBS = LOo; p.colAdd = pv2s; p.colBS = LOo;
        p.outF = A3; p.oFBS = (long long)LOo * LOo; p.ldoF = LOo;
        gemm(p, BC, 1);
    }
    k_softmax_ax1T<<<(BC * LOo + 7) / 8, 256, 0, stream>>>(A3, at3T, BC, LOo, LOo);
    { // OSK = att3^T @ OPK
        GemmP p = gp0();
        p.A = at3T; p.aBS = (long long)LOo * LOo; p.lda = LOo;
        p.B = OPKb; p.bBS = (long long)LOo * Hh; p.ldb = Hh;
        p.M = LOo; p.N = Hh; p.K = LOo;
        p.outF = OSK; p.oFBS = (long long)LOo * Hh; p.ldoF = Hh;
        gemm(p, BC, 0);
    }
    k_buildFcat<<<eblocks((long long)RO * Hh), 256, 0, stream>>>(OPKf, OSK, Fcat, (long long)RO * Hh);
    { // OFK = relu(Fcat @ Wf + b)
        GemmP p = gp0();
        p.A = Fcat; p.lda = 4 * Hh;
        p.B = wfT; p.ldb = 4 * Hh;
        p.M = RO; p.N = Hh; p.K = 4 * Hh;
        p.bias = Wfb; p.epi = 3;
        p.outF = OFK; p.ldoF = Hh;
        gemm(p, 1, 1);
    }
    k_maxred<<<(BC * Hh + 255) / 256, 256, 0, stream>>>(OFK, outp);
}
